// Qwen2MoELayer_54245436948772
// MI455X (gfx1250) — matrix operand order repaired and checked by reading, not compiled, not run on hardware
//
#include <hip/hip_runtime.h>
#include <hip/hip_bf16.h>
#include <math.h>

// ---------------------------------------------------------------------------
// Qwen2-style MoE layer for MI455X (gfx1250, wave32, WMMA bf16 + async->LDS)
// ---------------------------------------------------------------------------

typedef __attribute__((ext_vector_type(16))) __bf16 v16bf;
typedef __attribute__((ext_vector_type(8)))  float  v8f;

#define T_TOK   8192
#define H_DIM   2048
#define E_NUM   16
#define TOPK    4
#define IM      1408           // I_MOE
#define ISH     5632           // I_SHARED
#define MAX_TILES 2064         // T*K/16 + E (16-aligned padding per expert group)
#define PAD_ROWS  (MAX_TILES*16)

// --------------------------- fragment index helpers ------------------------
// A (bf16 16x32, MxK): lane L -> M=L%16, half=L/16.
//   VGPR v holds K = half*16 + 2v {,+1}.
__device__ __forceinline__ int a_kb(int v, int half) {
  return half * 16 + (v << 1);
}

// CDNA5 async Global->LDS copy (16B per active lane), tracked by ASYNCcnt.
__device__ __forceinline__ void async_ld_b128(unsigned lds_off, const void* gptr) {
  unsigned long long ga = (unsigned long long)(uintptr_t)gptr;
  asm volatile("global_load_async_to_lds_b128 %0, %1, off"
               :: "v"(lds_off), "v"(ga) : "memory");
}
__device__ __forceinline__ void wait_asynccnt0() {
  asm volatile("s_wait_asynccnt 0x0" ::: "memory");
}

// ------------------------------- init --------------------------------------
__global__ void init_kernel(int* counts, int* cursor, int* row_of_slot) {
  int i = blockIdx.x * blockDim.x + threadIdx.x;
  if (i < E_NUM) { counts[i] = 0; cursor[i] = 0; }
  if (i < PAD_ROWS) row_of_slot[i] = -1;
}

// ------------------------------- router ------------------------------------
__global__ void router_kernel(const float* __restrict__ hs,
                              const float* __restrict__ rw,
                              const float* __restrict__ segw,
                              float* __restrict__ topk_w, int* __restrict__ topk_idx,
                              float* __restrict__ gvec, int* __restrict__ counts) {
  __shared__ float sh[H_DIM];
  __shared__ float red[128];
  __shared__ float logit[E_NUM + 1];
  const int t = blockIdx.x, tid = threadIdx.x;
  for (int i = tid; i < H_DIM; i += 128) sh[i] = hs[(size_t)t * H_DIM + i];
  __syncthreads();
  for (int e = 0; e <= E_NUM; ++e) {
    const float* w = (e < E_NUM) ? (rw + (size_t)e * H_DIM) : segw;
    float p = 0.f;
    for (int i = tid; i < H_DIM; i += 128) p += sh[i] * w[i];
    red[tid] = p; __syncthreads();
    for (int s = 64; s > 0; s >>= 1) {
      if (tid < s) red[tid] += red[tid + s];
      __syncthreads();
    }
    if (tid == 0) logit[e] = red[0];
    __syncthreads();
  }
  if (tid == 0) {
    float mx = logit[0];
    for (int e = 1; e < E_NUM; ++e) mx = fmaxf(mx, logit[e]);
    float pr[E_NUM], s = 0.f;
    for (int e = 0; e < E_NUM; ++e) { pr[e] = expf(logit[e] - mx); s += pr[e]; }
    float inv = 1.f / s;
    for (int e = 0; e < E_NUM; ++e) pr[e] *= inv;
    for (int k = 0; k < TOPK; ++k) {
      int bi = 0; float bv = -1.f;
      for (int e = 0; e < E_NUM; ++e) if (pr[e] > bv) { bv = pr[e]; bi = e; }
      topk_w[t * TOPK + k] = bv;
      topk_idx[t * TOPK + k] = bi;
      atomicAdd(&counts[bi], 1);
      pr[bi] = -2.f;
    }
    gvec[t] = 1.f / (1.f + expf(-logit[E_NUM]));   // sigmoid shared-expert gate
  }
}

// ------------------------------- scan --------------------------------------
__global__ void scan_kernel(const int* __restrict__ counts, int* __restrict__ off_p,
                            int* __restrict__ tile_expert) {
  __shared__ int soff[E_NUM];
  __shared__ int spc[E_NUM];
  const int tid = threadIdx.x;
  if (tid == 0) {
    int run = 0;
    for (int e = 0; e < E_NUM; ++e) {
      soff[e] = run; off_p[e] = run;
      int a = (counts[e] + 15) & ~15;
      spc[e] = a; run += a;
    }
  }
  __syncthreads();
  for (int i = tid; i < MAX_TILES; i += blockDim.x) {
    int r = i * 16, e = -1;
    for (int q = 0; q < E_NUM; ++q)
      if (r >= soff[q] && r < soff[q] + spc[q]) { e = q; break; }
    tile_expert[i] = e;
  }
}

// ------------------------------- scatter -----------------------------------
__global__ void scatter_kernel(const int* __restrict__ topk_idx, const int* __restrict__ off_p,
                               int* __restrict__ cursor, int* __restrict__ row_of_slot,
                               int* __restrict__ slot_of) {
  int i = blockIdx.x * blockDim.x + threadIdx.x;
  if (i >= T_TOK * TOPK) return;
  int e = topk_idx[i];
  int pos = atomicAdd(&cursor[e], 1);
  int slot = off_p[e] + pos;
  row_of_slot[slot] = i / TOPK;
  slot_of[i] = slot;
}

// ---------------------- SwiGLU grouped GEMM (gate+up) ----------------------
// Block = 256 threads = 8 waves. One 16-row fp32 A tile per block, staged via
// CDNA5 async Global->LDS b128 copies issued by wave 0 (4 instrs = 2KB).
// LDA/LDB/LDC/KDIM are compile-time: all B loads become base+imm offsets.
template <int LDA, int LDB, int LDC, int KDIM>
__global__ void swiglu_gemm_kernel(const float* __restrict__ A, const int* __restrict__ row_map,
                                   const int* __restrict__ tile_expert,
                                   const float* __restrict__ Bg0, const float* __restrict__ Bu0,
                                   long long estride, __bf16* __restrict__ C) {
  __shared__ __attribute__((aligned(16))) float sAf[16][32];   // 2KB fp32 A tile
  __shared__ const float* rowp[16];
  const int rt = blockIdx.x;
  int e = 0;
  if (tile_expert) { e = tile_expert[rt]; if (e < 0) return; }  // uniform exit
  const int tid = threadIdx.x, lane = tid & 31, wid = tid >> 5;
  const int col0 = (blockIdx.y * 8 + wid) * 16;
  const int Mloc = lane & 15, half = lane >> 4;
  // Per-lane B base: column (col0+Mloc), K offset half*16 folded in.
  const float* Bg = Bg0 + (long long)e * estride + (size_t)half * 16 * LDB + col0 + Mloc;
  const float* Bu = Bu0 + (long long)e * estride + (size_t)half * 16 * LDB + col0 + Mloc;

  // Row base pointers (padding rows clamp to row 0: their outputs are never read)
  if (tid < 16) {
    int row = row_map ? row_map[rt * 16 + tid] : (rt * 16 + tid);
    if (row < 0) row = 0;
    rowp[tid] = A + (size_t)row * LDA;
  }
  __syncthreads();

  // Wave-0 per-lane async transfer descriptors: seg = i*32+lane covers
  // (row = seg/8, 16B chunk = seg%8) of the 16x32 fp32 tile.
  const float* gb[4];
  unsigned ldso[4];
  if (wid == 0) {
#pragma unroll
    for (int i = 0; i < 4; ++i) {
      int seg = i * 32 + lane;
      int r = seg >> 3, sc = seg & 7;
      gb[i] = rowp[r] + sc * 4;
      ldso[i] = (unsigned)(uintptr_t)(&sAf[r][sc * 4]);
    }
  }

  v8f accg = {}; v8f accu = {};
  for (int kk = 0; kk < KDIM; kk += 32) {
    __syncthreads();                      // WAR: prior iter's LDS reads done
    if (wid == 0) {
#pragma unroll
      for (int i = 0; i < 4; ++i) async_ld_b128(ldso[i], gb[i] + kk);
      wait_asynccnt0();
    }
    __syncthreads();                      // tile visible to all waves

    v16bf a, bgf, buf;
#pragma unroll
    for (int vv = 0; vv < 8; ++vv) {
      int kb = a_kb(vv, half);
      a[2 * vv]     = (__bf16)sAf[Mloc][kb];
      a[2 * vv + 1] = (__bf16)sAf[Mloc][kb + 1];
    }
    const float* pg = Bg + (size_t)kk * LDB;        // constant-imm offsets below
    const float* pu = Bu + (size_t)kk * LDB;
#pragma unroll
    for (int vv = 0; vv < 8; ++vv) {
      bgf[2 * vv]     = (__bf16)pg[(size_t)(2 * vv) * LDB];
      bgf[2 * vv + 1] = (__bf16)pg[(size_t)(2 * vv + 1) * LDB];
      buf[2 * vv]     = (__bf16)pu[(size_t)(2 * vv) * LDB];
      buf[2 * vv + 1] = (__bf16)pu[(size_t)(2 * vv + 1) * LDB];
    }
    accg = __builtin_amdgcn_wmma_f32_16x16x32_bf16(false, a, false, bgf, (short)0, accg, false, false);
    accu = __builtin_amdgcn_wmma_f32_16x16x32_bf16(false, a, false, buf, (short)0, accu, false, false);
  }
#pragma unroll
  for (int i = 0; i < 8; ++i) {                     // D: M = i + 8*half, N = lane%16
    float g = accg[i], u = accu[i];
    float s = (g / (1.f + expf(-g))) * u;           // silu(g)*u
    C[(size_t)(rt * 16 + i + 8 * half) * LDC + col0 + Mloc] = (__bf16)s;
  }
}

// ------------------- plain GEMM: bf16 activations x f32 W ------------------
// A tile is bf16 (1KB): wave 0 stages it with 2 async b128 instructions.
template <int LDA, int LDB, int LDC, int KDIM>
__global__ void gemm_bf16_kernel(const __bf16* __restrict__ A, const int* __restrict__ tile_expert,
                                 const float* __restrict__ B0, long long estride,
                                 const float* __restrict__ row_scale, float* __restrict__ C) {
  __shared__ __attribute__((aligned(16))) __bf16 sAh[16][32];  // 1KB bf16 A tile
  const int rt = blockIdx.x;
  int e = 0;
  if (tile_expert) { e = tile_expert[rt]; if (e < 0) return; }
  const int tid = threadIdx.x, lane = tid & 31, wid = tid >> 5;
  const int col0 = (blockIdx.y * 8 + wid) * 16;
  const int Mloc = lane & 15, half = lane >> 4;
  const float* B = B0 + (long long)e * estride + (size_t)half * 16 * LDB + col0 + Mloc;

  // seg = i*32+lane covers (row = seg/4, 16B chunk = seg%4) of the 16x32 bf16 tile
  const __bf16* gb[2];
  unsigned ldso[2];
  if (wid == 0) {
#pragma unroll
    for (int i = 0; i < 2; ++i) {
      int seg = i * 32 + lane;
      int r = seg >> 2, sc = seg & 3;
      gb[i] = A + (size_t)(rt * 16 + r) * LDA + sc * 8;
      ldso[i] = (unsigned)(uintptr_t)(&sAh[r][sc * 8]);
    }
  }

  v8f acc = {};
  for (int kk = 0; kk < KDIM; kk += 32) {
    __syncthreads();
    if (wid == 0) {
#pragma unroll
      for (int i = 0; i < 2; ++i) async_ld_b128(ldso[i], gb[i] + kk);
      wait_asynccnt0();
    }
    __syncthreads();

    v16bf a, b;
#pragma unroll
    for (int vv = 0; vv < 8; ++vv) {
      int kb = a_kb(vv, half);
      a[2 * vv]     = sAh[Mloc][kb];
      a[2 * vv + 1] = sAh[Mloc][kb + 1];
    }
    const float* pb = B + (size_t)kk * LDB;         // constant-imm offsets below
#pragma unroll
    for (int vv = 0; vv < 8; ++vv) {
      b[2 * vv]     = (__bf16)pb[(size_t)(2 * vv) * LDB];
      b[2 * vv + 1] = (__bf16)pb[(size_t)(2 * vv + 1) * LDB];
    }
    acc = __builtin_amdgcn_wmma_f32_16x16x32_bf16(false, a, false, b, (short)0, acc, false, false);
  }
#pragma unroll
  for (int i = 0; i < 8; ++i) {
    int row = rt * 16 + i + 8 * half;
    float sc = row_scale ? row_scale[row] : 1.f;
    C[(size_t)row * LDC + col0 + Mloc] = acc[i] * sc;
  }
}

// ------------------------------- combine -----------------------------------
__global__ void combine_kernel(const float* __restrict__ topk_w, const int* __restrict__ slot_of,
                               const float* __restrict__ expert_out, float* __restrict__ out) {
  const int t = blockIdx.x, tid = threadIdx.x;
  float w[TOPK]; int s[TOPK];
#pragma unroll
  for (int k = 0; k < TOPK; ++k) { w[k] = topk_w[t * TOPK + k]; s[k] = slot_of[t * TOPK + k]; }
  const float4* eo = (const float4*)expert_out;
  float4* o4 = (float4*)out;
  for (int n = tid; n < H_DIM / 4; n += blockDim.x) {
    float4 acc = o4[(size_t)t * (H_DIM / 4) + n];   // shared expert already here
#pragma unroll
    for (int k = 0; k < TOPK; ++k) {
      float4 v = eo[(size_t)s[k] * (H_DIM / 4) + n];
      acc.x += w[k] * v.x; acc.y += w[k] * v.y;
      acc.z += w[k] * v.z; acc.w += w[k] * v.w;
    }
    o4[(size_t)t * (H_DIM / 4) + n] = acc;
  }
}

// ------------------------------- launch ------------------------------------
extern "C" void kernel_launch(void* const* d_in, const int* in_sizes, int n_in,
                              void* d_out, int out_size, void* d_ws, size_t ws_size,
                              hipStream_t stream) {
  (void)in_sizes; (void)n_in; (void)out_size; (void)ws_size;
  const float* hs  = (const float*)d_in[0];   // [T,H]
  const float* rw  = (const float*)d_in[1];   // [E,H]
  const float* guw = (const float*)d_in[2];   // [E,H,2*IM]
  const float* dw  = (const float*)d_in[3];   // [E,IM,H]
  const float* sgw = (const float*)d_in[4];   // [H,ISH]
  const float* suw = (const float*)d_in[5];   // [H,ISH]
  const float* sdw = (const float*)d_in[6];   // [ISH,H]
  const float* seg = (const float*)d_in[7];   // [H,1]
  float* out = (float*)d_out;                 // [T,H] fp32

  char* p = (char*)d_ws;
  auto alloc = [&](size_t bytes) -> char* {
    char* q = p; p += (bytes + 255) & ~(size_t)255; return q;
  };
  int*   counts      = (int*)alloc(E_NUM * 4);
  int*   cursor      = (int*)alloc(E_NUM * 4);
  int*   off_p       = (int*)alloc(E_NUM * 4);
  int*   tile_expert = (int*)alloc(MAX_TILES * 4);
  float* topk_w      = (float*)alloc((size_t)T_TOK * TOPK * 4);
  int*   topk_idx    = (int*)alloc((size_t)T_TOK * TOPK * 4);
  float* gvec        = (float*)alloc((size_t)T_TOK * 4);
  int*   row_of_slot = (int*)alloc((size_t)PAD_ROWS * 4);
  int*   slot_of     = (int*)alloc((size_t)T_TOK * TOPK * 4);
  float* expert_out  = (float*)alloc((size_t)PAD_ROWS * H_DIM * 4);
  __bf16* act        = (__bf16*)alloc((size_t)PAD_ROWS * IM * 2); // aliased below
  __bf16* shared_act = act;  // MoE act is dead before the shared expert runs

  init_kernel<<<(PAD_ROWS + 255) / 256, 256, 0, stream>>>(counts, cursor, row_of_slot);
  router_kernel<<<T_TOK, 128, 0, stream>>>(hs, rw, seg, topk_w, topk_idx, gvec, counts);
  scan_kernel<<<1, 64, 0, stream>>>(counts, off_p, tile_expert);
  scatter_kernel<<<(T_TOK * TOPK + 255) / 256, 256, 0, stream>>>(
      topk_idx, off_p, cursor, row_of_slot, slot_of);

  // Routed experts: gate_up + SwiGLU  ->  act[PAD_ROWS, IM] (bf16)
  swiglu_gemm_kernel<H_DIM, 2 * IM, IM, H_DIM>
      <<<dim3(MAX_TILES, IM / 16 / 8), 256, 0, stream>>>(
          hs, row_of_slot, tile_expert, guw, guw + IM,
          (long long)H_DIM * 2 * IM, act);
  // Routed experts: down  ->  expert_out[PAD_ROWS, H] (f32)
  gemm_bf16_kernel<IM, H_DIM, H_DIM, IM>
      <<<dim3(MAX_TILES, H_DIM / 16 / 8), 256, 0, stream>>>(
          act, tile_expert, dw, (long long)IM * H_DIM, nullptr, expert_out);

  // Shared expert: gate/up + SwiGLU -> shared_act[T, ISH] (bf16)
  swiglu_gemm_kernel<H_DIM, ISH, ISH, H_DIM>
      <<<dim3(T_TOK / 16, ISH / 16 / 8), 256, 0, stream>>>(
          hs, nullptr, nullptr, sgw, suw, 0LL, shared_act);
  // Shared expert: down, scaled by sigmoid gate -> out directly
  gemm_bf16_kernel<ISH, H_DIM, H_DIM, ISH>
      <<<dim3(T_TOK / 16, H_DIM / 16 / 8), 256, 0, stream>>>(
          shared_act, nullptr, sdw, 0LL, gvec, out);

  // out += sum_k w[t,k] * expert_out[slot_of[t,k]]
  combine_kernel<<<T_TOK, 256, 0, stream>>>(topk_w, slot_of, expert_out, out);
}
